// DialogueGCN_4037269259075
// MI455X (gfx1250) — compile-verified, matrix operand order checked by reading, not run on hardware
//
#include <hip/hip_runtime.h>
#include <hip/hip_bf16.h>

// ---------------- problem constants ----------------
#define B_   64
#define L_   110
#define U_   768
#define G_   512
#define H_   256
#define FF_  2048
#define R_   8
#define NB_  30
#define TAG_ 7
#define N_   (B_*L_)
#define WIN_ 10

// ---------------- WMMA types ----------------
typedef __attribute__((ext_vector_type(16))) __bf16       v16bf;
typedef __attribute__((ext_vector_type(8)))  float        v8f;
typedef __attribute__((ext_vector_type(4)))  unsigned int v4u;

union FragBF { v16bf v; v4u u[2]; };

// =====================================================================
// Generic GEMM:  C(M,Nd) = A(M,K) @ W(Nd,K)^T  (+bias) (+relu)
// A,W bf16 row-major.  One wave computes a 32x64 tile (2 M-subtiles x
// 4 N-subtiles) with v_wmma_f32_16x16x32_bf16; B fragments are reused
// across the two M-subtiles (8 wmma per 12 b128 loads).  Fragments are
// register double-buffered: loads for k+1 issue before the wmmas for k,
// so waits are partial instead of full drains.
// block = 4 waves = 128 rows of M.
// mode: 0 = f32 store, 1 = bf16 store, 2 = f32 accumulate (+=)
// =====================================================================
__global__ __launch_bounds__(128)
void gemm_bf16_wmma(const __bf16* __restrict__ A, const __bf16* __restrict__ W,
                    const float* __restrict__ bias, void* __restrict__ Cout,
                    int M, int Nd, int K, int mode, int relu)
{
    const int lane = threadIdx.x & 31;
    const int wave = threadIdx.x >> 5;
    const int mrow = lane & 15;
    const int half = lane >> 4;          // 0: k 0-7,16-23 / rows 0-7, 1: k 8-15,24-31 / rows 8-15
    const int m0   = (blockIdx.y * 4 + wave) * 32;
    const int n0   = blockIdx.x * 64;
    if (m0 >= M) return;

    v8f acc[2][4];
#pragma unroll
    for (int mt = 0; mt < 2; ++mt)
#pragma unroll
        for (int t = 0; t < 4; ++t)
            acc[mt][t] = (v8f){0.f,0.f,0.f,0.f,0.f,0.f,0.f,0.f};

    const __bf16* arow0 = A + (size_t)(m0 + mrow) * K + 8 * half;
    const __bf16* arow1 = arow0 + (size_t)16 * K;
    const __bf16* wbase = W + (size_t)(n0 + mrow) * K + 8 * half;

    FragBF a0c, a1c, bc[4];
    // ---- prologue: load fragments for k = 0 ----
    a0c.u[0] = *(const v4u*)(arow0);      a0c.u[1] = *(const v4u*)(arow0 + 16);
    a1c.u[0] = *(const v4u*)(arow1);      a1c.u[1] = *(const v4u*)(arow1 + 16);
#pragma unroll
    for (int t = 0; t < 4; ++t) {
        const __bf16* wp = wbase + (size_t)(16 * t) * K;
        bc[t].u[0] = *(const v4u*)(wp);   bc[t].u[1] = *(const v4u*)(wp + 16);
    }

#pragma unroll 2
    for (int k0 = 32; k0 < K; k0 += 32) {
        // ---- issue loads for this k0 (consumed next round) ----
        FragBF a0n, a1n, bn[4];
        a0n.u[0] = *(const v4u*)(arow0 + k0);      a0n.u[1] = *(const v4u*)(arow0 + k0 + 16);
        a1n.u[0] = *(const v4u*)(arow1 + k0);      a1n.u[1] = *(const v4u*)(arow1 + k0 + 16);
#pragma unroll
        for (int t = 0; t < 4; ++t) {
            const __bf16* wp = wbase + (size_t)(16 * t) * K + k0;
            bn[t].u[0] = *(const v4u*)(wp);        bn[t].u[1] = *(const v4u*)(wp + 16);
        }
        // ---- compute on current fragments ----
#pragma unroll
        for (int t = 0; t < 4; ++t) {
            acc[0][t] = __builtin_amdgcn_wmma_f32_16x16x32_bf16(
                false, a0c.v, false, bc[t].v, (short)0, acc[0][t], false, false);
            acc[1][t] = __builtin_amdgcn_wmma_f32_16x16x32_bf16(
                false, a1c.v, false, bc[t].v, (short)0, acc[1][t], false, false);
        }
        // ---- rotate buffers ----
        a0c = a0n; a1c = a1n;
#pragma unroll
        for (int t = 0; t < 4; ++t) bc[t] = bn[t];
    }
    // ---- epilogue compute on last fragments ----
#pragma unroll
    for (int t = 0; t < 4; ++t) {
        acc[0][t] = __builtin_amdgcn_wmma_f32_16x16x32_bf16(
            false, a0c.v, false, bc[t].v, (short)0, acc[0][t], false, false);
        acc[1][t] = __builtin_amdgcn_wmma_f32_16x16x32_bf16(
            false, a1c.v, false, bc[t].v, (short)0, acc[1][t], false, false);
    }

#pragma unroll
    for (int t = 0; t < 4; ++t) {
        const int n  = n0 + 16 * t + mrow;
        const float bv = bias ? bias[n] : 0.f;
#pragma unroll
        for (int mt = 0; mt < 2; ++mt) {
#pragma unroll
            for (int r = 0; r < 8; ++r) {
                const int m = m0 + 16 * mt + r + 8 * half;
                float val = acc[mt][t][r] + bv;
                if (relu) val = fmaxf(val, 0.f);
                const size_t idx = (size_t)m * Nd + n;
                if (mode == 0)      ((float*)Cout)[idx] = val;
                else if (mode == 1) ((__bf16*)Cout)[idx] = (__bf16)val;
                else                ((float*)Cout)[idx] += val;
            }
        }
    }
}

// ---------------- conversions ----------------
__global__ void f32_to_bf16_kernel(const float* __restrict__ s, __bf16* __restrict__ d, size_t n)
{
    size_t i = (size_t)blockIdx.x * blockDim.x + threadIdx.x;
    if (i < n) d[i] = (__bf16)s[i];
}

// dst(cols,rows) bf16 = transpose of src(rows,cols) f32 ; batched over blockIdx.z
__global__ void transpose_conv_kernel(const float* __restrict__ src, __bf16* __restrict__ dst,
                                      int rows, int cols)
{
    src += (size_t)blockIdx.z * rows * cols;
    dst += (size_t)blockIdx.z * rows * cols;
    int i = blockIdx.x * blockDim.x + threadIdx.x;
    if (i < rows * cols) {
        int c = i / rows, r = i % rows;
        dst[(size_t)c * rows + r] = (__bf16)src[(size_t)r * cols + c];
    }
}

// w_rel[r,g,h] = sum_b comp[r,b] * basis[b,g,h]
__global__ void wrel_kernel(const float* __restrict__ comp, const float* __restrict__ basis,
                            float* __restrict__ wrel)
{
    int i = blockIdx.x * blockDim.x + threadIdx.x;
    if (i >= R_ * G_ * H_) return;
    int r = i / (G_ * H_), gh = i % (G_ * H_);
    float s = 0.f;
#pragma unroll 5
    for (int b = 0; b < NB_; ++b) s += comp[r * NB_ + b] * basis[(size_t)b * G_ * H_ + gh];
    wrel[i] = s;
}

// ---------------- single-head attention (per query row) ----------------
__global__ __launch_bounds__(128)
void attn_kernel(const __bf16* __restrict__ qkv, __bf16* __restrict__ out)
{
    const int row = blockIdx.x;          // b*L + j
    const int b   = row / L_;
    __shared__ float p[L_];
    __shared__ float inv_s;
    const float scale = 0.03608439182435161f;   // 1/sqrt(768)
    const int t = threadIdx.x;
    if (t < L_) {
        const __bf16* q  = qkv + (size_t)row * (3 * U_);
        const __bf16* kk = qkv + (size_t)(b * L_ + t) * (3 * U_) + U_;
        float s = 0.f;
        for (int d = 0; d < U_; ++d) s += (float)q[d] * (float)kk[d];
        p[t] = s * scale;
    }
    __syncthreads();
    if (t == 0) {
        float m = -1e30f;
        for (int i = 0; i < L_; ++i) m = fmaxf(m, p[i]);
        float sum = 0.f;
        for (int i = 0; i < L_; ++i) { p[i] = __expf(p[i] - m); sum += p[i]; }
        inv_s = 1.f / sum;
    }
    __syncthreads();
    const float inv = inv_s;
    const __bf16* vb = qkv + (size_t)(b * L_) * (3 * U_) + 2 * U_;
    for (int d = t; d < U_; d += 128) {
        float s = 0.f;
        for (int k = 0; k < L_; ++k) s += p[k] * (float)vb[(size_t)k * (3 * U_) + d];
        out[(size_t)row * U_ + d] = (__bf16)(s * inv);
    }
}

// ---------------- LayerNorm(a+b) -> optional f32 + bf16 ----------------
__global__ __launch_bounds__(256)
void ln_kernel(const float* __restrict__ xa, const float* __restrict__ xb,
               const float* __restrict__ g, const float* __restrict__ be,
               float* __restrict__ outf, __bf16* __restrict__ outh, int D)
{
    const int row = blockIdx.x;
    const float* a = xa + (size_t)row * D;
    const float* bb = xb + (size_t)row * D;
    __shared__ float sh[256];
    const int tid = threadIdx.x;
    float s = 0.f;
    for (int i = tid; i < D; i += 256) s += a[i] + bb[i];
    sh[tid] = s; __syncthreads();
    for (int o = 128; o > 0; o >>= 1) { if (tid < o) sh[tid] += sh[tid + o]; __syncthreads(); }
    const float mean = sh[0] / (float)D;
    __syncthreads();
    float v = 0.f;
    for (int i = tid; i < D; i += 256) { float d0 = a[i] + bb[i] - mean; v += d0 * d0; }
    sh[tid] = v; __syncthreads();
    for (int o = 128; o > 0; o >>= 1) { if (tid < o) sh[tid] += sh[tid + o]; __syncthreads(); }
    const float rstd = rsqrtf(sh[0] / (float)D + 1e-5f);
    for (int i = tid; i < D; i += 256) {
        float val = (a[i] + bb[i] - mean) * rstd * g[i] + be[i];
        if (outf) outf[(size_t)row * D + i] = val;
        outh[(size_t)row * D + i] = (__bf16)val;
    }
}

// ---------------- windowed edge-attention softmax (per source row j) ---
__global__ __launch_bounds__(128)
void alpha_kernel(const float* __restrict__ t, const __bf16* __restrict__ feats,
                  float* __restrict__ alpha)
{
    const int row = blockIdx.x;               // b*L + j
    const int b = row / L_, j = row % L_;
    const int klo = (j - WIN_ > 0) ? j - WIN_ : 0;
    const int khi = (j + WIN_ < L_ - 1) ? j + WIN_ : L_ - 1;
    __shared__ float sc[L_];
    __shared__ float inv_s;
    const int tt = threadIdx.x;
    if (tt >= klo && tt <= khi) {
        const float* tj = t + (size_t)row * G_;
        const __bf16* fk = feats + (size_t)(b * L_ + tt) * G_;
        float s = 0.f;
        for (int d = 0; d < G_; ++d) s += tj[d] * (float)fk[d];
        sc[tt] = s;
    }
    __syncthreads();
    if (tt == 0) {
        float m = -1e30f;
        for (int k = klo; k <= khi; ++k) m = fmaxf(m, sc[k]);
        float sum = 0.f;
        for (int k = klo; k <= khi; ++k) { sc[k] = __expf(sc[k] - m); sum += sc[k]; }
        inv_s = 1.f / sum;
    }
    __syncthreads();
    if (tt < L_) {
        float v = (tt >= klo && tt <= khi) ? sc[tt] * inv_s : 0.f;
        alpha[(size_t)row * L_ + tt] = v;
    }
}

// ---------------- RGCN per-(dst,relation) mean gather ------------------
// h1[dst] += mean_{j in window, etype==r} yr[b*L+j]
__global__ __launch_bounds__(256)
void rgcn_gather_kernel(const float* __restrict__ yr, float* __restrict__ h1,
                        const int* __restrict__ sp, int r)
{
    const int node = blockIdx.x;              // dst = b*L + k
    const int b = node / L_, k = node % L_;
    const int jlo = (k - WIN_ > 0) ? k - WIN_ : 0;
    const int jhi = (k + WIN_ < L_ - 1) ? k + WIN_ : L_ - 1;
    const int spd = sp[node];
    int cnt = 0;
    for (int j = jlo; j <= jhi; ++j) {
        int et = (sp[b * L_ + j] * 2 + spd) * 2 + ((j >= k) ? 1 : 0);
        if (et == r) cnt++;
    }
    if (cnt == 0) return;
    const int h = threadIdx.x;
    float s = 0.f;
    for (int j = jlo; j <= jhi; ++j) {
        int et = (sp[b * L_ + j] * 2 + spd) * 2 + ((j >= k) ? 1 : 0);
        if (et == r) s += yr[(size_t)(b * L_ + j) * H_ + h];
    }
    h1[(size_t)node * H_ + h] += s / (float)cnt;
}

// ---------------- GraphConv weighted gather ---------------------------
__global__ __launch_bounds__(256)
void gconv_gather_kernel(const float* __restrict__ h1, const float* __restrict__ alpha,
                         float* __restrict__ agg)
{
    const int node = blockIdx.x;              // dst = b*L + k
    const int b = node / L_, k = node % L_;
    const int jlo = (k - WIN_ > 0) ? k - WIN_ : 0;
    const int jhi = (k + WIN_ < L_ - 1) ? k + WIN_ : L_ - 1;
    const int h = threadIdx.x;
    float s = 0.f;
    for (int j = jlo; j <= jhi; ++j) {
        float w = alpha[(size_t)(b * L_ + j) * L_ + k];
        s += w * h1[(size_t)(b * L_ + j) * H_ + h];
    }
    agg[(size_t)node * H_ + h] = s;
}

// ---------------- concat [feats_bf16 | bf16(h2)] ----------------------
__global__ void concat_kernel(const __bf16* __restrict__ feats, const float* __restrict__ h2,
                              __bf16* __restrict__ z)
{
    int i = blockIdx.x * blockDim.x + threadIdx.x;
    if (i >= N_ * (G_ + H_)) return;
    int n = i / (G_ + H_), c = i % (G_ + H_);
    z[i] = (c < G_) ? feats[(size_t)n * G_ + c] : (__bf16)h2[(size_t)n * H_ + (c - G_)];
}

// ---------------- tiny classifier tail (H=256 -> 7) -------------------
__global__ void clf2_kernel(const __bf16* __restrict__ c1, const float* __restrict__ w,
                            const float* __restrict__ b2, float* __restrict__ out)
{
    int i = blockIdx.x * blockDim.x + threadIdx.x;
    if (i >= N_ * TAG_) return;
    int n = i / TAG_, tg = i % TAG_;
    float s = b2[tg];
    const __bf16* cp = c1 + (size_t)n * H_;
    const float*  wp = w + (size_t)tg * H_;
    for (int h = 0; h < H_; ++h) s += (float)cp[h] * wp[h];
    out[i] = s;
}

// =====================================================================
extern "C" void kernel_launch(void* const* d_in, const int* in_sizes, int n_in,
                              void* d_out, int out_size, void* d_ws, size_t ws_size,
                              hipStream_t stream)
{
    const float* x        = (const float*)d_in[0];
    const float* inpw     = (const float*)d_in[1];
    const float* inpb     = (const float*)d_in[2];
    const float* outw     = (const float*)d_in[3];
    const float* outb     = (const float*)d_in[4];
    const float* ln1g     = (const float*)d_in[5];
    const float* ln1b     = (const float*)d_in[6];
    const float* ff1w     = (const float*)d_in[7];
    const float* ff1b     = (const float*)d_in[8];
    const float* ff2w     = (const float*)d_in[9];
    const float* ff2b     = (const float*)d_in[10];
    const float* ln2g     = (const float*)d_in[11];
    const float* ln2b     = (const float*)d_in[12];
    const float* tow      = (const float*)d_in[13];
    const float* tob      = (const float*)d_in[14];
    const float* attw     = (const float*)d_in[15];
    const float* basis    = (const float*)d_in[16];
    const float* comp     = (const float*)d_in[17];
    const float* root     = (const float*)d_in[18];
    const float* rgcnb    = (const float*)d_in[19];
    const float* gcrelw   = (const float*)d_in[20];
    const float* gcrelb   = (const float*)d_in[21];
    const float* gcrootw  = (const float*)d_in[22];
    const float* clf1w    = (const float*)d_in[23];
    const float* clf1b    = (const float*)d_in[24];
    const float* clf2w    = (const float*)d_in[25];
    const float* clf2b    = (const float*)d_in[26];
    const int*   speaker  = (const int*)d_in[27];
    float* out = (float*)d_out;

    // --------------- workspace layout (bump allocator, 256B aligned) ----
    char* ws = (char*)d_ws;
    size_t off = 0;
    auto alloc = [&](size_t bytes) { size_t o = off; off += (bytes + 255) & ~(size_t)255; return o; };

    const size_t W_INP   = alloc((size_t)3 * U_ * U_ * 2);
    const size_t W_OUT   = alloc((size_t)U_ * U_ * 2);
    const size_t W_FF1   = alloc((size_t)FF_ * U_ * 2);
    const size_t W_FF2   = alloc((size_t)U_ * FF_ * 2);
    const size_t W_TO    = alloc((size_t)G_ * U_ * 2);
    const size_t W_ATTT  = alloc((size_t)G_ * G_ * 2);
    const size_t W_ROOTT = alloc((size_t)H_ * G_ * 2);
    const size_t W_WRELT = alloc((size_t)R_ * H_ * G_ * 2);
    const size_t W_GCREL = alloc((size_t)H_ * H_ * 2);
    const size_t W_GCRT  = alloc((size_t)H_ * H_ * 2);
    const size_t W_CLF1  = alloc((size_t)H_ * (G_ + H_) * 2);
    const size_t WRELF   = alloc((size_t)R_ * G_ * H_ * 4);
    const size_t FEATSBF = alloc((size_t)N_ * G_ * 2);
    const size_t U1      = alloc((size_t)N_ * (3 * U_) * 2);   // qkvbf -> ffbuf -> zcat/c1/h1bf/aggbf
    const size_t U2      = alloc((size_t)N_ * U_ * 2);         // xbf -> abf -> hbf
    const size_t U3      = alloc((size_t)N_ * U_ * 4);         // aout/h -> tbuf -> agg,h2
    const size_t U4      = alloc((size_t)N_ * U_ * 4);         // ff2out -> alpha,yr,h1
    (void)ws_size;

    __bf16* XBF   = (__bf16*)(ws + U2);
    __bf16* ABF   = (__bf16*)(ws + U2);
    __bf16* HBF   = (__bf16*)(ws + U2);
    __bf16* QKVBF = (__bf16*)(ws + U1);
    __bf16* FFBUF = (__bf16*)(ws + U1);
    __bf16* ZCAT  = (__bf16*)(ws + U1);
    __bf16* C1BF  = (__bf16*)(ws + U1 + (size_t)N_ * (G_ + H_) * 2);
    __bf16* H1BF  = (__bf16*)(ws + U1 + 14417920);
    __bf16* AGGBF = (__bf16*)(ws + U1 + 18022400);
    float*  AOUT  = (float*)(ws + U3);
    float*  TBUF  = (float*)(ws + U3);
    float*  AGG   = (float*)(ws + U3);
    float*  H2    = (float*)(ws + U3 + (size_t)N_ * H_ * 4);
    float*  FF2O  = (float*)(ws + U4);
    float*  ALPHA = (float*)(ws + U4);
    float*  YR    = (float*)(ws + U4 + 3097600);
    float*  H1    = (float*)(ws + U4 + 3097600 + (size_t)N_ * H_ * 4);
    float*  WREL  = (float*)(ws + WRELF);

    auto conv = [&](const float* s, __bf16* d, size_t n) {
        f32_to_bf16_kernel<<<dim3((unsigned)((n + 255) / 256)), 256, 0, stream>>>(s, d, n);
    };
    auto gemm = [&](const __bf16* A, const __bf16* W, const float* bias, void* C,
                    int M, int Nd, int K, int mode, int relu) {
        gemm_bf16_wmma<<<dim3(Nd / 64, (M + 127) / 128), 128, 0, stream>>>(A, W, bias, C, M, Nd, K, mode, relu);
    };

    // ---- phase 0: weight/input conversions ----
    conv(x,       XBF,                 (size_t)N_ * U_);
    conv(inpw,    (__bf16*)(ws + W_INP),  (size_t)3 * U_ * U_);
    conv(outw,    (__bf16*)(ws + W_OUT),  (size_t)U_ * U_);
    conv(ff1w,    (__bf16*)(ws + W_FF1),  (size_t)FF_ * U_);
    conv(ff2w,    (__bf16*)(ws + W_FF2),  (size_t)U_ * FF_);
    conv(tow,     (__bf16*)(ws + W_TO),   (size_t)G_ * U_);
    conv(gcrelw,  (__bf16*)(ws + W_GCREL),(size_t)H_ * H_);
    conv(gcrootw, (__bf16*)(ws + W_GCRT), (size_t)H_ * H_);
    conv(clf1w,   (__bf16*)(ws + W_CLF1), (size_t)H_ * (G_ + H_));
    transpose_conv_kernel<<<dim3((G_*G_ + 255)/256, 1, 1), 256, 0, stream>>>(attw, (__bf16*)(ws + W_ATTT), G_, G_);
    transpose_conv_kernel<<<dim3((G_*H_ + 255)/256, 1, 1), 256, 0, stream>>>(root, (__bf16*)(ws + W_ROOTT), G_, H_);
    wrel_kernel<<<dim3((R_*G_*H_ + 255)/256), 256, 0, stream>>>(comp, basis, WREL);
    transpose_conv_kernel<<<dim3((G_*H_ + 255)/256, 1, R_), 256, 0, stream>>>(WREL, (__bf16*)(ws + W_WRELT), G_, H_);

    // ---- phase 1: SeqContext transformer ----
    gemm(XBF, (__bf16*)(ws + W_INP), inpb, QKVBF, N_, 3 * U_, U_, 1, 0);
    attn_kernel<<<dim3(N_), 128, 0, stream>>>(QKVBF, ABF);
    gemm(ABF, (__bf16*)(ws + W_OUT), outb, AOUT, N_, U_, U_, 0, 0);
    ln_kernel<<<dim3(N_), 256, 0, stream>>>(x, AOUT, ln1g, ln1b, AOUT, HBF, U_);      // h (f32 in-place) + hbf
    gemm(HBF, (__bf16*)(ws + W_FF1), ff1b, FFBUF, N_, FF_, U_, 1, 1);                 // relu(h@ff1^T)
    gemm(FFBUF, (__bf16*)(ws + W_FF2), ff2b, FF2O, N_, U_, FF_, 0, 0);
    ln_kernel<<<dim3(N_), 256, 0, stream>>>(AOUT, FF2O, ln2g, ln2b, nullptr, HBF, U_); // hseq -> bf16 only
    gemm(HBF, (__bf16*)(ws + W_TO), tob, (__bf16*)(ws + FEATSBF), N_, G_, U_, 1, 0);   // feats (bf16)

    // ---- phase 2: EdgeAtt ----
    gemm((__bf16*)(ws + FEATSBF), (__bf16*)(ws + W_ATTT), nullptr, TBUF, N_, G_, G_, 0, 0); // t = feats@att_w
    alpha_kernel<<<dim3(N_), 128, 0, stream>>>(TBUF, (__bf16*)(ws + FEATSBF), ALPHA);

    // ---- phase 3: RGCN ----
    gemm((__bf16*)(ws + FEATSBF), (__bf16*)(ws + W_ROOTT), rgcnb, H1, N_, H_, G_, 0, 0);    // xf@root + bias
    for (int r = 0; r < R_; ++r) {
        gemm((__bf16*)(ws + FEATSBF), (__bf16*)(ws + W_WRELT) + (size_t)r * H_ * G_,
             nullptr, YR, N_, H_, G_, 0, 0);                                                // y_r = xf@w_rel[r]
        rgcn_gather_kernel<<<dim3(N_), H_, 0, stream>>>(YR, H1, speaker, r);
    }

    // ---- phase 4: GraphConv ----
    gconv_gather_kernel<<<dim3(N_), H_, 0, stream>>>(H1, ALPHA, AGG);
    conv(H1,  H1BF,  (size_t)N_ * H_);
    conv(AGG, AGGBF, (size_t)N_ * H_);
    gemm(AGGBF, (__bf16*)(ws + W_GCREL), gcrelb, H2, N_, H_, H_, 0, 0);
    gemm(H1BF,  (__bf16*)(ws + W_GCRT),  nullptr, H2, N_, H_, H_, 2, 0);                    // accumulate

    // ---- phase 5: classifier ----
    concat_kernel<<<dim3((N_ * (G_ + H_) + 255) / 256), 256, 0, stream>>>((__bf16*)(ws + FEATSBF), H2, ZCAT);
    gemm(ZCAT, (__bf16*)(ws + W_CLF1), clf1b, C1BF, N_, H_, G_ + H_, 1, 1);                 // relu
    clf2_kernel<<<dim3((N_ * TAG_ + 255) / 256), 256, 0, stream>>>(C1BF, clf2w, clf2b, out);
}
